// SDT_BiGRU_Classifier_63007170233044
// MI455X (gfx1250) — matrix operand order repaired and checked by reading, not compiled, not run on hardware
//
#include <hip/hip_runtime.h>

// ---------------------------------------------------------------------------
// Shapes (fixed by the reference): B=4096, N=25, D=256, H=8, K=3
// BN = B*N = 102400 rows, divisible by 16 (6400 row tiles).
// ---------------------------------------------------------------------------
#define BN      102400
#define DMODEL  256
#define NNODE   25
#define NHEAD   8
#define KCONV   3
#define KIM2COL 768          // 3*256 im2col depth
#define NK_CONV 24           // 768/32 k-steps
#define NK_OUT  8            // 256/32 k-steps

typedef __attribute__((ext_vector_type(16))) __bf16 v16bf;
typedef __attribute__((ext_vector_type(8)))  float  v8f;

// A-matrix (16-bit, 16x32) fragment K index for vector element j, lane-half hi.
// Per ISA table: VGPR0..3 hold K=0..7 (lanes 0-15) / K=8..15 (lanes 16-31),
// VGPR4..7 hold K=16..23 / K=24..31.
__device__ __forceinline__ int ka_map(int j, int hi) {
  return (j < 8 ? j : j + 8) + hi * 8;
}
// B-matrix (16-bit, 32x16) fragment K index: same order as A (lanes 0-15 hold
// K=0..7,16..23; lanes 16-31 hold K=8..15,24..31).
__device__ __forceinline__ int kb_map(int j, int hi) {
  return (j < 8 ? j : j + 8) + hi * 8;
}

// ---------------------------------------------------------------------------
// Weight packing: conv weights -> im2col B fragments.
// wc layout: [proj(3)][jt(16)][kk(24)][lane(32)][j(16)] bf16
// im2col row index kg = t*256 + din ;  B[kg][dout] = w[dout][din][t]
// ---------------------------------------------------------------------------
__global__ void pack_wc(const float* __restrict__ wq,
                        const float* __restrict__ wk,
                        const float* __restrict__ wv,
                        __bf16* __restrict__ wc) {
  int idx = blockIdx.x * 256 + threadIdx.x;
  if (idx >= 3 * 16 * NK_CONV * 32 * 16) return;
  int j = idx & 15;
  int t1 = idx >> 4;
  int lane = t1 & 31; t1 >>= 5;
  int kk = t1 % NK_CONV; t1 /= NK_CONV;
  int jt = t1 & 15;
  int p  = t1 >> 4;
  int hi = lane >> 4, nidx = lane & 15;
  int dout = jt * 16 + nidx;
  int kg = kk * 32 + kb_map(j, hi);
  int t = kg >> 8, din = kg & 255;
  const float* w = (p == 0) ? wq : (p == 1) ? wk : wv;
  wc[idx] = (__bf16)w[(dout * DMODEL + din) * KCONV + t];
}

// wo packing: out = a @ wo^T  =>  B[din][dout] = wo[dout][din]
// wop layout: [jt(16)][kk(8)][lane(32)][j(16)] bf16
__global__ void pack_wo(const float* __restrict__ wo, __bf16* __restrict__ wop) {
  int idx = blockIdx.x * 256 + threadIdx.x;
  if (idx >= 16 * NK_OUT * 32 * 16) return;
  int j = idx & 15;
  int t1 = idx >> 4;
  int lane = t1 & 31; t1 >>= 5;
  int kk = t1 & 7;
  int jt = t1 >> 3;
  int hi = lane >> 4, nidx = lane & 15;
  int dout = jt * 16 + nidx;
  int din = kk * 32 + kb_map(j, hi);
  wop[idx] = (__bf16)wo[dout * DMODEL + din];
}

// ---------------------------------------------------------------------------
// Fused conv-QKV (WMMA GEMM over im2col) + LayerNorm + residual -> bf16 q,k,v
// Block: 512 threads = 16 waves; each block owns 16 rows of [BN, 256].
// Wave w computes output columns [16w, 16w+16) for all three projections.
// ---------------------------------------------------------------------------
__global__ __launch_bounds__(512) void conv_ln_kernel(
    const float* __restrict__ x, const __bf16* __restrict__ wc,
    const float* __restrict__ gq, const float* __restrict__ bq,
    const float* __restrict__ gk, const float* __restrict__ bk,
    const float* __restrict__ gv, const float* __restrict__ bv,
    __bf16* __restrict__ qb, __bf16* __restrict__ kb, __bf16* __restrict__ vb) {
  __shared__ __align__(32) __bf16 Atile[NK_CONV * 32 * 16];  // 24 KB
  __shared__ float red[3][16][2];                            // per-row sum, sumsq

  const int tid = threadIdx.x;
  if (tid < 96) ((float*)red)[tid] = 0.0f;

  // ---- Phase A: build im2col A tile in fragment order -------------------
  {
    const int lane16 = tid >> 4;       // 0..31 (acts as WMMA lane)
    const int j      = tid & 15;
    const int hi     = lane16 >> 4;
    const int M      = lane16 & 15;
    const int row    = blockIdx.x * 16 + M;
    const int bb     = row / NNODE;
    const int n      = row % NNODE;
#pragma unroll
    for (int kk = 0; kk < NK_CONV; ++kk) {
      int kg = kk * 32 + ka_map(j, hi);
      int t = kg >> 8, din = kg & 255;
      int np = n + t - 1;
      float v = (np >= 0 && np < NNODE)
                    ? x[((size_t)(bb * NNODE + np)) * DMODEL + din]
                    : 0.0f;
      Atile[(kk * 32 + lane16) * 16 + j] = (__bf16)v;
    }
  }
  __syncthreads();

  // ---- Phase B: WMMA GEMM, 3 projections share the A fragments ----------
  const int wvid = tid >> 5;   // wave id == jt (output col tile)
  const int lane = tid & 31;
  v8f accs[3] = {};
  const v16bf* Ab = (const v16bf*)Atile;
  for (int kk = 0; kk < NK_CONV; ++kk) {
    v16bf a = Ab[kk * 32 + lane];
#pragma unroll
    for (int p = 0; p < 3; ++p) {
      v16bf bf = *(const v16bf*)(wc + ((size_t)(((p * 16 + wvid) * NK_CONV + kk) * 32 + lane) << 4));
      accs[p] = __builtin_amdgcn_wmma_f32_16x16x32_bf16(
          false, a, false, bf, (short)0, accs[p], false, false);
    }
  }

  // ---- Phase C: fused LayerNorm + residual ------------------------------
  const int hiw  = lane >> 4;
  const int nidx = lane & 15;
  const int col  = wvid * 16 + nidx;

#pragma unroll
  for (int p = 0; p < 3; ++p) {
#pragma unroll
    for (int r = 0; r < 8; ++r) {
      float val = accs[p][r];
      float s = val, q2 = val * val;
#pragma unroll
      for (int mlt = 1; mlt < 16; mlt <<= 1) {
        s  += __shfl_xor(s, mlt, 32);
        q2 += __shfl_xor(q2, mlt, 32);
      }
      if (nidx == 0) {
        int Mr = r + 8 * hiw;
        atomicAdd(&red[p][Mr][0], s);
        atomicAdd(&red[p][Mr][1], q2);
      }
    }
  }
  __syncthreads();

  const float* gp[3] = {gq, gk, gv};
  const float* bp[3] = {bq, bk, bv};
  __bf16* outp[3] = {qb, kb, vb};

  float xr[8];
#pragma unroll
  for (int r = 0; r < 8; ++r) {
    int Mr = r + 8 * hiw;
    xr[r] = x[((size_t)(blockIdx.x * 16 + Mr)) * DMODEL + col];
  }
#pragma unroll
  for (int p = 0; p < 3; ++p) {
    float gg = gp[p][col];
    float bbv = bp[p][col];
#pragma unroll
    for (int r = 0; r < 8; ++r) {
      int Mr = r + 8 * hiw;
      float mean = red[p][Mr][0] * (1.0f / 256.0f);
      float var  = red[p][Mr][1] * (1.0f / 256.0f) - mean * mean;
      float rstd = rsqrtf(var + 1e-5f);
      float o = (accs[p][r] - mean) * rstd * gg + bbv + xr[r];
      outp[p][((size_t)(blockIdx.x * 16 + Mr)) * DMODEL + col] = (__bf16)o;
    }
  }
}

// ---------------------------------------------------------------------------
// Attention: one wave per (batch, head). N=25, dk=32.
// Lane m owns score row m; K/V staged in LDS per wave.
// ---------------------------------------------------------------------------
__global__ __launch_bounds__(256) void attn_kernel(
    const __bf16* __restrict__ qb, const __bf16* __restrict__ kb,
    const __bf16* __restrict__ vb, const float* __restrict__ rel_table,
    const float* __restrict__ local_bias, const unsigned char* __restrict__ adj,
    __bf16* __restrict__ ob) {
  __shared__ float KV[8][2][NNODE * 32];  // 51.2 KB
  const int wvid = threadIdx.x >> 5;
  const int lane = threadIdx.x & 31;
  const int pair = blockIdx.x * 8 + wvid;
  const int b = pair >> 3;
  const int h = pair & 7;

  float* Kl = KV[wvid][0];
  float* Vl = KV[wvid][1];
  for (int i = lane; i < NNODE * 32; i += 32) {
    int mm = i >> 5, d = i & 31;
    size_t base = ((size_t)(b * NNODE + mm)) * DMODEL + h * 32 + d;
    Kl[i] = (float)kb[base];
    Vl[i] = (float)vb[base];
  }
  __syncthreads();

  const int m = lane;
  if (m < NNODE) {
    float qr[32];
#pragma unroll
    for (int d = 0; d < 32; ++d)
      qr[d] = (float)qb[((size_t)(b * NNODE + m)) * DMODEL + h * 32 + d];

    float sc[NNODE];
    float mx = -3.0e38f;
#pragma unroll
    for (int mm = 0; mm < NNODE; ++mm) {
      float s = 0.0f;
#pragma unroll
      for (int d = 0; d < 32; ++d) s += qr[d] * Kl[mm * 32 + d];
      s *= 0.17677669529663687f;  // 1/sqrt(32)
      s += rel_table[(m - mm + NNODE - 1) * NHEAD + h];
      s += local_bias[(h * NNODE + m) * NNODE + mm];
      s = (adj[m * NNODE + mm] != 0) ? s : -3.0e38f;
      sc[mm] = s;
      mx = fmaxf(mx, s);
    }
    float sum = 0.0f;
#pragma unroll
    for (int mm = 0; mm < NNODE; ++mm) {
      sc[mm] = __expf(sc[mm] - mx);
      sum += sc[mm];
    }
    float inv = 1.0f / sum;
#pragma unroll
    for (int d = 0; d < 32; ++d) {
      float o = 0.0f;
#pragma unroll
      for (int mm = 0; mm < NNODE; ++mm) o += sc[mm] * Vl[mm * 32 + d];
      ob[((size_t)(b * NNODE + m)) * DMODEL + h * 32 + d] = (__bf16)(o * inv);
    }
  }
}

// ---------------------------------------------------------------------------
// Output projection: out = attn_out @ wo^T + bo   (WMMA, fp32 out)
// ---------------------------------------------------------------------------
__global__ __launch_bounds__(512) void oproj_kernel(
    const __bf16* __restrict__ ob, const __bf16* __restrict__ wop,
    const float* __restrict__ bo, float* __restrict__ out) {
  __shared__ __align__(32) __bf16 Atile[NK_OUT * 32 * 16];  // 8 KB

  const int tid = threadIdx.x;
  {
    const int lane16 = tid >> 4;
    const int j      = tid & 15;
    const int hi     = lane16 >> 4;
    const int M      = lane16 & 15;
    const size_t row = (size_t)blockIdx.x * 16 + M;
#pragma unroll
    for (int kk = 0; kk < NK_OUT; ++kk) {
      int kg = kk * 32 + ka_map(j, hi);
      Atile[(kk * 32 + lane16) * 16 + j] = ob[row * DMODEL + kg];
    }
  }
  __syncthreads();

  const int wvid = tid >> 5;
  const int lane = tid & 31;
  v8f acc = {};
  const v16bf* Ab = (const v16bf*)Atile;
#pragma unroll
  for (int kk = 0; kk < NK_OUT; ++kk) {
    v16bf a = Ab[kk * 32 + lane];
    v16bf bf = *(const v16bf*)(wop + ((size_t)((wvid * NK_OUT + kk) * 32 + lane) << 4));
    acc = __builtin_amdgcn_wmma_f32_16x16x32_bf16(
        false, a, false, bf, (short)0, acc, false, false);
  }

  const int hiw  = lane >> 4;
  const int nidx = lane & 15;
  const int col  = wvid * 16 + nidx;
  const float bov = bo[col];
#pragma unroll
  for (int r = 0; r < 8; ++r) {
    int Mr = r + 8 * hiw;
    out[((size_t)(blockIdx.x * 16 + Mr)) * DMODEL + col] = acc[r] + bov;
  }
}

// ---------------------------------------------------------------------------
extern "C" void kernel_launch(void* const* d_in, const int* in_sizes, int n_in,
                              void* d_out, int out_size, void* d_ws, size_t ws_size,
                              hipStream_t stream) {
  const float* x   = (const float*)d_in[0];
  const unsigned char* adj = (const unsigned char*)d_in[1];
  const float* wq  = (const float*)d_in[2];
  const float* wk  = (const float*)d_in[3];
  const float* wvw = (const float*)d_in[4];
  const float* gq  = (const float*)d_in[5];
  const float* bq  = (const float*)d_in[6];
  const float* gk  = (const float*)d_in[7];
  const float* bk  = (const float*)d_in[8];
  const float* gv  = (const float*)d_in[9];
  const float* bv  = (const float*)d_in[10];
  const float* rel = (const float*)d_in[11];
  const float* lb  = (const float*)d_in[12];
  const float* wo  = (const float*)d_in[13];
  const float* bo  = (const float*)d_in[14];

  const size_t BND = (size_t)BN * DMODEL;
  __bf16* qb  = (__bf16*)d_ws;
  __bf16* kb  = qb + BND;
  __bf16* vb  = kb + BND;
  __bf16* ob  = vb + BND;
  __bf16* wcp = ob + BND;                       // 3*16*24*32*16 = 589824 elems
  __bf16* wop = wcp + 3 * 16 * NK_CONV * 32 * 16;  // 65536 elems

  pack_wc<<<(3 * 16 * NK_CONV * 32 * 16 + 255) / 256, 256, 0, stream>>>(wq, wk, wvw, wcp);
  pack_wo<<<(16 * NK_OUT * 32 * 16 + 255) / 256, 256, 0, stream>>>(wo, wop);
  conv_ln_kernel<<<BN / 16, 512, 0, stream>>>(x, wcp, gq, bq, gk, bk, gv, bv, qb, kb, vb);
  attn_kernel<<<(BN / NNODE) * NHEAD / 8, 256, 0, stream>>>(qb, kb, vb, rel, lb, adj, ob);
  oproj_kernel<<<BN / 16, 512, 0, stream>>>(ob, wop, bo, (float*)d_out);
}
